// AttnConv2d_77214922047686
// MI455X (gfx1250) — matrix operand order repaired and checked by reading, not compiled, not run on hardware
//
#include <hip/hip_runtime.h>

// AttnConv2d pipeline for MI455X (gfx1250, wave32, WMMA), round 3:
//  - conv B-operand staged in LDS TRANSPOSED (K/ci fastest, padded stride 264
//    halfs = 528B so half-wave lanes hit distinct 16B bank slots) ->
//    B fragment = 2x ds_load_b128 at immediate offsets (was 16x ds_load_u16)
//  - each wave computes 16co x 32px: one A fragment feeds two WMMAs
//  - K reordered r-major (kg = r*64 + ci): compile-time (kh,kw) per K-chunk
//  - scores GEMM: affine pointer stepping (+288 elems per K-chunk)

#define CIN   64
#define COUT  64
#define HH    96
#define WW    96
#define NPIX  (HH * WW)         // 9216
#define KDIM  (CIN * 9)         // 576
#define NB    16
#define LDSN  (3 * CIN * 34)    // 6528 staged halfs per tile
#define DXS   264               // padded halfs per dx slice (>= 3*64=192)

typedef __attribute__((ext_vector_type(16))) _Float16 v16h;
typedef __attribute__((ext_vector_type(8)))  _Float16 v8h;
typedef __attribute__((ext_vector_type(8)))  float    v8f;

static __device__ __forceinline__ v16h cat8(v8h lo, v8h hi) {
  v16h r;
#pragma unroll
  for (int j = 0; j < 8; ++j) { r[j] = lo[j]; r[8 + j] = hi[j]; }
  return r;
}

// ------------------------------------------------- weights cvt + K-permute
// wh[z][co][r*64+ci] = w_z[co][ci][kh][kw],  r = kh*3+kw
__global__ void cvt_weights(const float* __restrict__ w1,
                            const float* __restrict__ w2,
                            const float* __restrict__ w3,
                            _Float16* __restrict__ wh) {
  int i = blockIdx.x * blockDim.x + threadIdx.x;
  if (i >= 3 * COUT * KDIM) return;
  int z = i / (COUT * KDIM);
  int t = i % (COUT * KDIM);
  int co = t / KDIM;
  int k  = t % KDIM;
  int r  = k >> 6;
  int ci = k & 63;
  const float* src = (z == 0) ? w1 : (z == 1) ? w2 : w3;
  wh[i] = (_Float16)src[co * KDIM + ci * 9 + r];
}

// --------------------------------------------------- conv (f32 in -> f16 out)
// grid: (96 rows * 3 col-segs, B, 3 convs); block 128 = 4 waves.
// Workgroup computes 64co x 32px; wave wy: 16co x 32px (2 accumulators).
__global__ __launch_bounds__(128) void conv3x3_f32_f16(
    const float* __restrict__ x1, const float* __restrict__ x2,
    const _Float16* __restrict__ wh,
    _Float16* __restrict__ kfh, _Float16* __restrict__ qfh,
    _Float16* __restrict__ vfh) {
  const int z   = blockIdx.z;
  const int b   = blockIdx.y;
  const int py  = blockIdx.x / 3;
  const int px0 = (blockIdx.x % 3) * 32;
  const float*    X = ((z == 1) ? x2 : x1) + (size_t)b * CIN * NPIX;
  const _Float16* W = wh + (size_t)z * COUT * KDIM;
  _Float16*       O = ((z == 0) ? kfh : (z == 1) ? qfh : vfh) +
                      (size_t)b * COUT * NPIX;

  __shared__ _Float16 xs[34 * DXS];    // [dx:34][dy:3][ci:64] (ci fastest)

  // ---- stage halo tile: coalesced global reads, transposed LDS writes
  for (int i = threadIdx.x; i < LDSN; i += 128) {
    int dy  = i / (CIN * 34);
    int rem = i - dy * (CIN * 34);
    int ci  = rem / 34;
    int dx  = rem - ci * 34;
    int yy  = py + dy - 1;
    int xx  = px0 + dx - 1;
    float v = 0.0f;
    if ((unsigned)yy < (unsigned)HH && (unsigned)xx < (unsigned)WW)
      v = X[ci * NPIX + yy * WW + xx];
    xs[dx * DXS + dy * CIN + ci] = (_Float16)v;
  }
  __syncthreads();

  const int lane = threadIdx.x & 31;
  const int wy   = threadIdx.x >> 5;   // co strip (0..3)
  const int hh   = lane >> 4;
  const int l15  = lane & 15;

  const _Float16* Wrow = W + (size_t)(wy * 16 + l15) * KDIM + hh * 16;
  const _Float16* bp0  = xs + l15 * DXS + 16 * hh;        // px strip 0..15
  const _Float16* bp1  = bp0 + 16 * DXS;                  // px strip 16..31

  v8f acc0 = {}, acc1 = {};
#pragma unroll
  for (int kk = 0; kk < KDIM; kk += 32) {
    const int r   = kk >> 6;           // compile-time after unroll
    const int cib = kk & 63;
    const int kh  = r / 3, kw = r % 3;
    const int off = kw * DXS + kh * CIN + cib;            // compile-time

    v16h a  = cat8(*(const v8h*)(Wrow + kk), *(const v8h*)(Wrow + kk + 8));
    v16h b0 = cat8(*(const v8h*)(bp0 + off), *(const v8h*)(bp0 + off + 8));
    v16h b1 = cat8(*(const v8h*)(bp1 + off), *(const v8h*)(bp1 + off + 8));

    acc0 = __builtin_amdgcn_wmma_f32_16x16x32_f16(false, a, false, b0,
                                                  (short)0, acc0, false, false);
    acc1 = __builtin_amdgcn_wmma_f32_16x16x32_f16(false, a, false, b1,
                                                  (short)0, acc1, false, false);
  }

#pragma unroll
  for (int r = 0; r < 8; ++r) {
    int co = wy * 16 + r + 8 * hh;
    O[(size_t)co * NPIX + py * WW + px0 + l15]      = (_Float16)acc0[r];
    O[(size_t)co * NPIX + py * WW + px0 + 16 + l15] = (_Float16)acc1[r];
  }
}

// ------------------------------------------- dynamic conv (f16 in -> f32 out)
__global__ __launch_bounds__(128) void dynconv3x3_f16_f32(
    const _Float16* __restrict__ vfh, const _Float16* __restrict__ attnh,
    float* __restrict__ out) {
  const int b   = blockIdx.y;
  const int py  = blockIdx.x / 3;
  const int px0 = (blockIdx.x % 3) * 32;
  const _Float16* X = vfh + (size_t)b * COUT * NPIX;
  const _Float16* W = attnh + (size_t)b * COUT * KDIM;  // per-batch weights
  float*          O = out + (size_t)b * COUT * NPIX;

  __shared__ _Float16 xs[34 * DXS];

  for (int i = threadIdx.x; i < LDSN; i += 128) {
    int dy  = i / (CIN * 34);
    int rem = i - dy * (CIN * 34);
    int ci  = rem / 34;
    int dx  = rem - ci * 34;
    int yy  = py + dy - 1;
    int xx  = px0 + dx - 1;
    _Float16 v = (_Float16)0.0f;
    if ((unsigned)yy < (unsigned)HH && (unsigned)xx < (unsigned)WW)
      v = X[ci * NPIX + yy * WW + xx];
    xs[dx * DXS + dy * CIN + ci] = v;
  }
  __syncthreads();

  const int lane = threadIdx.x & 31;
  const int wy   = threadIdx.x >> 5;
  const int hh   = lane >> 4;
  const int l15  = lane & 15;

  const _Float16* Wrow = W + (size_t)(wy * 16 + l15) * KDIM + hh * 16;
  const _Float16* bp0  = xs + l15 * DXS + 16 * hh;
  const _Float16* bp1  = bp0 + 16 * DXS;

  v8f acc0 = {}, acc1 = {};
#pragma unroll
  for (int kk = 0; kk < KDIM; kk += 32) {
    const int r   = kk >> 6;
    const int cib = kk & 63;
    const int kh  = r / 3, kw = r % 3;
    const int off = kw * DXS + kh * CIN + cib;

    v16h a  = cat8(*(const v8h*)(Wrow + kk), *(const v8h*)(Wrow + kk + 8));
    v16h b0 = cat8(*(const v8h*)(bp0 + off), *(const v8h*)(bp0 + off + 8));
    v16h b1 = cat8(*(const v8h*)(bp1 + off), *(const v8h*)(bp1 + off + 8));

    acc0 = __builtin_amdgcn_wmma_f32_16x16x32_f16(false, a, false, b0,
                                                  (short)0, acc0, false, false);
    acc1 = __builtin_amdgcn_wmma_f32_16x16x32_f16(false, a, false, b1,
                                                  (short)0, acc1, false, false);
  }

#pragma unroll
  for (int r = 0; r < 8; ++r) {
    int co = wy * 16 + r + 8 * hh;
    O[(size_t)co * NPIX + py * WW + px0 + l15]      = acc0[r];
    O[(size_t)co * NPIX + py * WW + px0 + 16 + l15] = acc1[r];
  }
}

// ------------------------------------------------------------- scores (K^T Q)
// grid: (9 taps, B); block 512 = 16 waves, one 16x16 tile each.
__global__ __launch_bounds__(512) void scores_wmma(
    const _Float16* __restrict__ kfh, const _Float16* __restrict__ qfh,
    float* __restrict__ sbuf) {
  const int kidx = blockIdx.x;
  const int b    = blockIdx.y;
  const int kh   = kidx / 3, kw = kidx % 3;
  const int lane = threadIdx.x & 31;
  const int wid  = threadIdx.x >> 5;
  const int wy   = wid >> 2, wx = wid & 3;
  const int hh   = lane >> 4, l15 = lane & 15;

  const int cA = wy * 16 + l15;
  const int dB = wx * 16 + l15;
  const _Float16* Kp = kfh + (size_t)b * COUT * NPIX +
                       (size_t)cA * NPIX + kh * WW + kw + 48 * hh;
  const _Float16* Qp = qfh + (size_t)b * COUT * NPIX +
                       (size_t)dB * NPIX + kh * WW + kw + 48 * hh;

  v8f acc = {};
  for (int it = 0; it < 32; ++it) {
    v16h a, bf;
#pragma unroll
    for (int j = 0; j < 16; ++j) {
      a[j]  = Kp[3 * j];
      bf[j] = Qp[3 * j];
    }
    acc = __builtin_amdgcn_wmma_f32_16x16x32_f16(false, a, false, bf,
                                                 (short)0, acc, false, false);
    Kp += 288;
    Qp += 288;
  }
#pragma unroll
  for (int r = 0; r < 8; ++r) {
    int c = wy * 16 + r + 8 * hh;
    int d = wx * 16 + l15;
    sbuf[(size_t)(b * COUT + d) * KDIM + c * 9 + kidx] = acc[r] * (1.0f / 24.0f);
  }
}

// ------------------------------------------------------------------- softmax
// Reads sbuf rows (order c*9+r), writes attn rows permuted to r*64+c.
__global__ __launch_bounds__(128) void softmax576(const float* __restrict__ s,
                                                  _Float16* __restrict__ attn) {
  const int row = blockIdx.x;                      // b*64 + d
  const float* sr = s + (size_t)row * KDIM;
  _Float16*    ar = attn + (size_t)row * KDIM;
  __shared__ float red[128];
  const int t = threadIdx.x;

  float m = -3.4e38f;
  for (int i = t; i < KDIM; i += 128) m = fmaxf(m, sr[i]);
  red[t] = m; __syncthreads();
  for (int off = 64; off > 0; off >>= 1) {
    if (t < off) red[t] = fmaxf(red[t], red[t + off]);
    __syncthreads();
  }
  m = red[0]; __syncthreads();

  float sum = 0.0f;
  for (int i = t; i < KDIM; i += 128) sum += __expf(sr[i] - m);
  red[t] = sum; __syncthreads();
  for (int off = 64; off > 0; off >>= 1) {
    if (t < off) red[t] += red[t + off];
    __syncthreads();
  }
  const float inv = 1.0f / red[0];
  for (int i = t; i < KDIM; i += 128) {
    int c = i / 9, r = i - c * 9;
    ar[r * 64 + c] = (_Float16)(__expf(sr[i] - m) * inv);
  }
}

// -------------------------------------------------------------------- launch
extern "C" void kernel_launch(void* const* d_in, const int* in_sizes, int n_in,
                              void* d_out, int out_size, void* d_ws,
                              size_t ws_size, hipStream_t stream) {
  const float* x1 = (const float*)d_in[0];
  const float* x2 = (const float*)d_in[1];
  const float* w1 = (const float*)d_in[2];
  const float* w2 = (const float*)d_in[3];
  const float* w3 = (const float*)d_in[4];
  float* out = (float*)d_out;

  _Float16* wh    = (_Float16*)d_ws;                         // [3][64][576]
  _Float16* kfh   = wh + (size_t)3 * COUT * KDIM;            // [16][64][9216]
  _Float16* qfh   = kfh + (size_t)NB * COUT * NPIX;
  _Float16* vfh   = qfh + (size_t)NB * COUT * NPIX;
  float*    sbuf  = (float*)(vfh + (size_t)NB * COUT * NPIX);// [16][64][576]
  _Float16* attnh = (_Float16*)(sbuf + (size_t)NB * COUT * KDIM);

  cvt_weights<<<dim3((3 * COUT * KDIM + 255) / 256), 256, 0, stream>>>(
      w1, w2, w3, wh);
  conv3x3_f32_f16<<<dim3(HH * 3, NB, 3), 128, 0, stream>>>(
      x1, x2, wh, kfh, qfh, vfh);
  scores_wmma<<<dim3(9, NB), 512, 0, stream>>>(kfh, qfh, sbuf);
  softmax576<<<dim3(NB * COUT), 128, 0, stream>>>(sbuf, attnh);
  dynconv3x3_f16_f32<<<dim3(HH * 3, NB), 128, 0, stream>>>(vfh, attnh, out);
}
